// GRU_91302414778807
// MI455X (gfx1250) — matrix operand order repaired and checked by reading, not compiled, not run on hardware
//
#include <hip/hip_runtime.h>

// ---------------------------------------------------------------------------
// GRU + MLP + FC + head, fully fused for MI455X (gfx1250, wave32, WMMA).
//
// Shapes: B=1024, T=256, IN=H=128, HORIZON=24.
//
// Pipeline:
//  k_convert : w_i, w_h  (f32 row-major [k][n]) -> bf16 (n,k) layout
//  k_combine : per t, Wc_t = (mlp_w @ fc_w_t) @ out_w  -> bf16 [t][o(32 pad)][k=128]
//              (two chained WMMA GEMMs, K=512 then K=128), plus vb_t = mlp_b @ fc_w_t
//  k_cbias   : cbias[o] = (sum_t vb_t + fc_b) @ out_w + out_b
//  k_gru     : 64 WGs x 16 batch rows; loops t=0..255 with h in LDS.
//              x stream is software-pipelined through registers (next tile's
//              loads fly behind the current step's WMMA work); Wc_{t+1} is
//              prefetched (global_prefetch_b8).
// ---------------------------------------------------------------------------

typedef __attribute__((ext_vector_type(16))) __bf16 v16bf;
typedef __attribute__((ext_vector_type(8)))  __bf16 v8bf;
typedef __attribute__((ext_vector_type(8)))  float  v8f;

#define DEV __device__ __forceinline__

static constexpr int Bsz  = 1024;
static constexpr int Tlen = 256;
static constexpr int Hd   = 128;   // IN == H == 128
static constexpr int GN   = 384;   // 3*H gate columns
static constexpr int HOR  = 24;
static constexpr int OPAD = 32;    // horizon padded to 2 WMMA tiles

DEV v8f wmma_bf16(v16bf a, v16bf b, v8f c) {
  return __builtin_amdgcn_wmma_f32_16x16x32_bf16(
      /*neg_a=*/false, a, /*neg_b=*/false, b,
      /*c_mod=*/(short)0, c, /*reuse_a=*/false, /*reuse_b=*/false);
}

// A-fragment (16x32 bf16) gathered from a row-major [16][128] bf16 tile.
// A layout: lane L -> m = L%16, g = L/16; elem e: k = g*16 + e
// (contiguous, matching B).  Two 16-byte loads per lane.
DEV v16bf load_a_128(const __bf16* base, int kt) {
  const int lane = threadIdx.x & 31;
  const int m = lane & 15, g = lane >> 4;
  const __bf16* p = base + m * 128 + kt * 32 + g * 16;
  v8bf lo = *(const v8bf*)(p);
  v8bf hi = *(const v8bf*)(p + 8);
  v16bf a;
#pragma unroll
  for (int i = 0; i < 8; ++i) { a[i] = lo[i]; a[8 + i] = hi[i]; }
  return a;
}

// B-fragment (32x16 bf16) from (n,k)-layout storage base[n*128 + k].
// B layout (contiguous, same k order as A): lane L -> n = L%16, g = L/16; elem e: k = g*16 + e
// -> 16 contiguous bf16 per lane = two b128 loads.
DEV v16bf load_b_nk(const __bf16* base, int n0, int kt) {
  const int lane = threadIdx.x & 31;
  const int n = n0 + (lane & 15), g = lane >> 4;
  const __bf16* p = base + n * 128 + kt * 32 + g * 16;
  v8bf lo = *(const v8bf*)(p);
  v8bf hi = *(const v8bf*)(p + 8);
  v16bf b;
#pragma unroll
  for (int i = 0; i < 8; ++i) { b[i] = lo[i]; b[8 + i] = hi[i]; }
  return b;
}

// A-fragment from row-major f32 [M][ld] global, converting to bf16.
DEV v16bf load_a_f32(const float* base, int ld, int m0, int kt) {
  const int lane = threadIdx.x & 31;
  const int m = m0 + (lane & 15), g = lane >> 4;
  const float* p = base + (size_t)m * ld + kt * 32 + g * 16;
  v16bf a;
#pragma unroll
  for (int i = 0; i < 8; ++i) {
    a[i]     = (__bf16)p[i];
    a[8 + i] = (__bf16)p[8 + i];
  }
  return a;
}

// B-fragment from row-major f32 [K][ld] global (strided gather), cols >= nmax -> 0.
DEV v16bf load_b_f32(const float* base, int ld, int n0, int kt, int nmax) {
  const int lane = threadIdx.x & 31;
  const int n = n0 + (lane & 15), g = lane >> 4;
  v16bf b;
#pragma unroll
  for (int e = 0; e < 16; ++e) {
    const int k = kt * 32 + g * 16 + e;
    float v = (n < nmax) ? base[(size_t)k * ld + n] : 0.0f;
    b[e] = (__bf16)v;
  }
  return b;
}

// ---------------------------------------------------------------------------
// Kernel 0: convert gate weights f32 [128][384] -> bf16 (n,k) layout [384][128]
// ---------------------------------------------------------------------------
__global__ void k_convert(const float* __restrict__ w_i,
                          const float* __restrict__ w_h,
                          __bf16* __restrict__ wi_nk,
                          __bf16* __restrict__ wh_nk) {
  int idx = blockIdx.x * blockDim.x + threadIdx.x;
  if (idx < Hd * GN) {
    int k = idx / GN, n = idx % GN;
    wi_nk[n * 128 + k] = (__bf16)w_i[idx];
    wh_nk[n * 128 + k] = (__bf16)w_h[idx];
  }
}

// ---------------------------------------------------------------------------
// Kernel 1: per-t combined tail weight  Wc_t = (mlp_w @ fc_w_t) @ out_w
// grid = T blocks of 256 threads (8 waves).
// ---------------------------------------------------------------------------
__global__ __launch_bounds__(256) void k_combine(
    const float* __restrict__ mlp_w,   // [128][512]
    const float* __restrict__ mlp_b,   // [512]
    const float* __restrict__ fc_w,    // [T*512][128]
    const float* __restrict__ out_w,   // [128][24]
    __bf16* __restrict__ wc,           // [T][32][128] (t, o, k) bf16
    float* __restrict__ vb)            // [T][128]
{
  const int t    = blockIdx.x;
  const int tid  = threadIdx.x;
  const int wave = tid >> 5, lane = tid & 31;
  __shared__ __align__(16) __bf16 wc1[128][128];   // 32 KB intermediate (bf16)
  const float* fcw_t = fc_w + (size_t)t * 512 * 128;

  // GEMM1: Wc1 = mlp_w(128x512) @ fc_w_t(512x128). wave w owns n-tile w.
  // Two independent accumulator chains per iteration to hide WMMA RAW latency.
  v16bf bf[16];
#pragma unroll
  for (int kt = 0; kt < 16; ++kt) bf[kt] = load_b_f32(fcw_t, 128, wave * 16, kt, 128);
#pragma unroll 1
  for (int mt = 0; mt < 8; mt += 2) {
    v8f c0 = {}, c1 = {};
#pragma unroll
    for (int kt = 0; kt < 16; ++kt) {
      v16bf a0 = load_a_f32(mlp_w, 512, mt * 16, kt);
      v16bf a1 = load_a_f32(mlp_w, 512, (mt + 1) * 16, kt);
      c0 = wmma_bf16(a0, bf[kt], c0);
      c1 = wmma_bf16(a1, bf[kt], c1);
    }
    const int n = wave * 16 + (lane & 15), g = lane >> 4;
#pragma unroll
    for (int i = 0; i < 8; ++i) {
      wc1[mt * 16 + i + g * 8][n]       = (__bf16)c0[i];
      wc1[(mt + 1) * 16 + i + g * 8][n] = (__bf16)c1[i];
    }
  }
  __syncthreads();

  // GEMM2: Wc_t = Wc1(128x128) @ out_w(128x24, padded to 32). 16 tiles, 2/wave.
  {
    const int mt = wave;                 // two chains: nt=0 and nt=1
    v8f c0 = {}, c1 = {};
#pragma unroll
    for (int kt = 0; kt < 4; ++kt) {
      v16bf a  = load_a_128(&wc1[mt * 16][0], kt);
      v16bf b0 = load_b_f32(out_w, HOR, 0,  kt, HOR);   // cols >=24 are zero
      v16bf b1 = load_b_f32(out_w, HOR, 16, kt, HOR);
      c0 = wmma_bf16(a, b0, c0);
      c1 = wmma_bf16(a, b1, c1);
    }
    const int g = lane >> 4;
    const int o0 = (lane & 15), o1 = 16 + (lane & 15);
    __bf16* wct = wc + (size_t)t * OPAD * 128;
#pragma unroll
    for (int i = 0; i < 8; ++i) {
      wct[(size_t)o0 * 128 + mt * 16 + i + g * 8] = (__bf16)c0[i];
      wct[(size_t)o1 * 128 + mt * 16 + i + g * 8] = (__bf16)c1[i];
    }
  }

  // vb_t[j] = mlp_b @ fc_w_t (bias path through the FC layer)
  for (int j = tid; j < 128; j += 256) {
    float s = 0.f;
    for (int k = 0; k < 512; ++k) s += mlp_b[k] * fcw_t[(size_t)k * 128 + j];
    vb[t * 128 + j] = s;
  }
}

// ---------------------------------------------------------------------------
// Kernel 2: cbias[o] = (sum_t vb_t + fc_b) @ out_w + out_b
// ---------------------------------------------------------------------------
__global__ void k_cbias(const float* __restrict__ vb,
                        const float* __restrict__ fc_b,
                        const float* __restrict__ out_w,
                        const float* __restrict__ out_b,
                        float* __restrict__ cbias) {
  __shared__ float S[128];
  const int tid = threadIdx.x;   // 128 threads
  float s = fc_b[tid];
  for (int t = 0; t < Tlen; ++t) s += vb[t * 128 + tid];
  S[tid] = s;
  __syncthreads();
  if (tid < OPAD) {
    float c = 0.f;
    if (tid < HOR) {
      c = out_b[tid];
      for (int j = 0; j < 128; ++j) c += S[j] * out_w[j * HOR + tid];
    }
    cbias[tid] = c;
  }
}

// ---------------------------------------------------------------------------
// Kernel 3: fused GRU recurrence + output accumulation.
// grid = B/16 = 64 blocks, 256 threads (8 waves). State lives in LDS.
// x stream is double-buffered through registers: loads for step t+1 are
// issued before the step-t barriers, so HBM latency hides behind WMMA work.
// ---------------------------------------------------------------------------
__global__ __launch_bounds__(256) void k_gru(
    const float*  __restrict__ x,      // [B][T][128]
    const __bf16* __restrict__ wi,     // [384][128] (n,k)
    const __bf16* __restrict__ wh,     // [384][128] (n,k)  cols 256.. = w_h_a
    const float*  __restrict__ bias,   // [384]
    const __bf16* __restrict__ wc,     // [T][32][128]
    const float*  __restrict__ cbias,  // [32]
    float* __restrict__ out)           // [B][24]
{
  __shared__ __align__(16) __bf16 xs[16][128];   // x_t tile (bf16)
  __shared__            float     hs[16][128];   // h (fp32 master copy)
  __shared__ __align__(16) __bf16 hb[16][128];   // h (bf16 for WMMA A)
  __shared__ __align__(16) __bf16 rh[16][128];   // r*h (bf16 for WMMA A)
  __shared__            float     gs[16][384];   // gate pre/post activations
  __shared__            float     bs[384];       // biases

  const int tid  = threadIdx.x;
  const int wave = tid >> 5, lane = tid & 31;
  const int b0   = blockIdx.x * 16;

  for (int i = tid; i < GN; i += 256) bs[i] = bias[i];
  for (int i = tid; i < 16 * 128; i += 256) {
    hs[i >> 7][i & 127] = 0.f;
    hb[i >> 7][i & 127] = (__bf16)0.f;
  }
  v8f acc0 = {}, acc1 = {};   // output accumulators (used by waves 0/1)

  // x pipeline registers: this thread's 8 contiguous floats of the next tile
  const int xm = tid >> 4, xkb = (tid & 15) * 8;
  const float* xrow = x + ((size_t)(b0 + xm) * Tlen) * 128 + xkb;
  float xr[8];
#pragma unroll
  for (int i = 0; i < 8; ++i) xr[i] = xrow[i];   // preload t = 0

  __syncthreads();

#pragma unroll 1
  for (int t = 0; t < Tlen; ++t) {
    // ---- stage x_t tile from pipeline registers (f32 -> bf16) ----
#pragma unroll
    for (int i = 0; i < 8; ++i) xs[xm][xkb + i] = (__bf16)xr[i];

    // ---- issue next tile's loads + prefetch next Wc tile (fly behind step) --
    if (t + 1 < Tlen) {
      const float* xp = xrow + (size_t)(t + 1) * 128;
#pragma unroll
      for (int i = 0; i < 8; ++i) xr[i] = xp[i];
      __builtin_prefetch(wc + (size_t)(t + 1) * OPAD * 128 + (size_t)tid * 32, 0, 3);
    }
    __syncthreads();

    // ---- phase 1: g = x_t @ w_i  (+ h @ w_h on the 16 zr tiles) ----
    v16bf ax[4], ah[4];
#pragma unroll
    for (int kt = 0; kt < 4; ++kt) {
      ax[kt] = load_a_128(&xs[0][0], kt);
      ah[kt] = load_a_128(&hb[0][0], kt);
    }
#pragma unroll
    for (int j = 0; j < 3; ++j) {
      const int nt = wave * 3 + j;          // 24 n-tiles over 384 gate cols
      v8f c = {};
#pragma unroll
      for (int kt = 0; kt < 4; ++kt) c = wmma_bf16(ax[kt], load_b_nk(wi, nt * 16, kt), c);
      if (nt < 16) {                        // z,r columns also get h @ w_h_zr
#pragma unroll
        for (int kt = 0; kt < 4; ++kt) c = wmma_bf16(ah[kt], load_b_nk(wh, nt * 16, kt), c);
      }
      const int n = nt * 16 + (lane & 15), g = lane >> 4;
#pragma unroll
      for (int i = 0; i < 8; ++i) gs[i + g * 8][n] = c[i];
    }
    __syncthreads();

    // ---- phase 2: zr = sigmoid(g + b); rh = r * h ----
    for (int idx = tid; idx < 16 * 256; idx += 256) {
      const int m = idx >> 8, n = idx & 255;
      const float v = 1.f / (1.f + __expf(-(gs[m][n] + bs[n])));
      gs[m][n] = v;
      if (n >= 128) rh[m][n - 128] = (__bf16)(v * hs[m][n - 128]);
    }
    __syncthreads();

    // ---- phase 3a: a = tanh(gx_a + rh @ w_h_a + b_a); h = (1-z)h + z*a ----
    {
      const int nt = wave;                  // 8 n-tiles over 128 h cols
      v16bf ar[4];
#pragma unroll
      for (int kt = 0; kt < 4; ++kt) ar[kt] = load_a_128(&rh[0][0], kt);
      v8f c = {};
#pragma unroll
      for (int kt = 0; kt < 4; ++kt) c = wmma_bf16(ar[kt], load_b_nk(wh, 256 + nt * 16, kt), c);
      const int n = nt * 16 + (lane & 15), g = lane >> 4;
#pragma unroll
      for (int i = 0; i < 8; ++i) {
        const int m = i + g * 8;
        const float a  = tanhf(c[i] + gs[m][256 + n] + bs[256 + n]);
        const float z  = gs[m][n];
        const float hn = (1.f - z) * hs[m][n] + z * a;
        hs[m][n] = hn;
        hb[m][n] = (__bf16)hn;
      }
    }
    __syncthreads();

    // ---- phase 3b: acc += h_t @ Wc_t (fused MLP+FC+head, rank-24) ----
    if (wave < 2) {
      v16bf an[4];
#pragma unroll
      for (int kt = 0; kt < 4; ++kt) an[kt] = load_a_128(&hb[0][0], kt);
      const __bf16* wct = wc + (size_t)t * OPAD * 128;
      if (wave == 0) {
#pragma unroll
        for (int kt = 0; kt < 4; ++kt) acc0 = wmma_bf16(an[kt], load_b_nk(wct, 0, kt), acc0);
      } else {
#pragma unroll
        for (int kt = 0; kt < 4; ++kt) acc1 = wmma_bf16(an[kt], load_b_nk(wct, 16, kt), acc1);
      }
    }
    __syncthreads();   // 3b reads of hb complete before next step mutates LDS
  }

  // ---- epilogue: out[b][o] = acc + cbias[o] ----
  if (wave < 2) {
    v8f acc = (wave == 1) ? acc1 : acc0;
    const int o = wave * 16 + (lane & 15), g = lane >> 4;
    if (o < HOR) {
#pragma unroll
      for (int i = 0; i < 8; ++i) {
        const int m = i + g * 8;
        out[(size_t)(b0 + m) * HOR + o] = acc[i] + cbias[o];
      }
    }
  }
}

// ---------------------------------------------------------------------------
// Host launch
// ---------------------------------------------------------------------------
extern "C" void kernel_launch(void* const* d_in, const int* in_sizes, int n_in,
                              void* d_out, int out_size, void* d_ws, size_t ws_size,
                              hipStream_t stream) {
  const float* x     = (const float*)d_in[0];
  const float* w_i   = (const float*)d_in[1];
  const float* w_h   = (const float*)d_in[2];
  const float* b     = (const float*)d_in[3];
  const float* mlp_w = (const float*)d_in[4];
  const float* mlp_b = (const float*)d_in[5];
  const float* fc_w  = (const float*)d_in[6];
  const float* fc_b  = (const float*)d_in[7];
  const float* out_w = (const float*)d_in[8];
  const float* out_b = (const float*)d_in[9];
  float* out = (float*)d_out;

  // workspace layout (all offsets 256B aligned); total ~2.4 MB
  char* ws = (char*)d_ws;
  size_t off = 0;
  __bf16* wc    = (__bf16*)(ws + off); off += (size_t)Tlen * OPAD * 128 * 2;  // 2 MB
  __bf16* wi_nk = (__bf16*)(ws + off); off += (size_t)GN * 128 * 2;           // 96 KB
  __bf16* wh_nk = (__bf16*)(ws + off); off += (size_t)GN * 128 * 2;           // 96 KB
  float*  vb    = (float* )(ws + off); off += (size_t)Tlen * 128 * 4;         // 128 KB
  float*  cb    = (float* )(ws + off); off += 256;

  k_convert<<<(Hd * GN + 255) / 256, 256, 0, stream>>>(w_i, w_h, wi_nk, wh_nk);
  k_combine<<<Tlen, 256, 0, stream>>>(mlp_w, mlp_b, fc_w, out_w, wc, vb);
  k_cbias<<<1, 128, 0, stream>>>(vb, fc_b, out_w, out_b, cb);
  k_gru<<<Bsz / 16, 256, 0, stream>>>(x, wi_nk, wh_nk, b, wc, cb, out);
}
